// SwarmNeighborhoodEncoderAttention_8813272891532
// MI455X (gfx1250) — compile-verified, matrix operand order checked by reading, not run on hardware
//
#include <hip/hip_runtime.h>
#include <hip/hip_bf16.h>

// SwarmNeighborhoodEncoderAttention, fully fused for MI455X (gfx1250, wave32).
// One wave per batch element; K=16 neighbors == one 16-row WMMA tile, so the
// mean / softmax / weighted-sum are all tile-local. All GEMMs use
// v_wmma_f32_16x16x32_f16. Activations are kept in LDS in A-fragment packed
// layout (fragment load = 2x ds_load_b128); packed-B weights stream
// sequentially from L2 with one-block software pipelining + global_prefetch.

typedef __attribute__((ext_vector_type(16))) _Float16 v16h;
typedef __attribute__((ext_vector_type(2)))  _Float16 v2h;
typedef __attribute__((ext_vector_type(8)))  float    v8f;

#define WMMA_F16(a, b, c) \
  __builtin_amdgcn_wmma_f32_16x16x32_f16(false, (a), false, (b), (short)0, (c), false, false)

__device__ static inline float gelu_exact(float x) {
  return 0.5f * x * (1.0f + erff(x * 0.70710678118654752f));
}

// ---- A-fragment packed layout ------------------------------------------------
// Fragment for k-tile kt lives at buf[(kt*32 + lane)*16 .. +15]; lane/element
// mapping follows the CDNA5 16-bit A 16x32 layout:
//   lane = m + 16*h',  h' = (k%16)/8,  e = 2*(4*((k%32)/16) + ((k%8)/2)) + (k%2)

// Contiguous A-fragment load (2x ds_load_b128 per lane).
__device__ static inline v16h packed_frag(const _Float16* buf, int kt) {
  const int lane = threadIdx.x & 31;
  return *(const v16h*)(buf + ((size_t)(kt * 32 + lane)) * 16);
}

// Store a 16x16 f32 D-tile (lane holds (m = vv + 8*(lane>>4), n = lane&15))
// into the packed A-fragment buffer at global column k = nbase + (lane&15).
__device__ static inline void store_packed(_Float16* buf, int nbase, v8f acc, bool gelu) {
  const int lane = threadIdx.x & 31;
  const int k    = nbase + (lane & 15);
  const int kt   = k >> 5;
  const int ko   = k & 31;
  const int rem  = ko & 15;
  const int e    = 2 * (((ko & 16) >> 2) + ((rem & 7) >> 1)) + (rem & 1);
  const int dl0  = ((rem >> 3) << 4) + ((lane >> 4) << 3);  // dest lane for vv=0
  _Float16* p = buf + ((kt * 32 + dl0) * 16 + e);
#pragma unroll
  for (int vv = 0; vv < 8; ++vv) {             // m = mb + vv -> dest lane +1 -> +16 halves
    float x = acc[vv];
    if (gelu) x = gelu_exact(x);
    p[vv * 16] = (_Float16)x;
  }
}

// Gather-style A-fragment (used only for the broadcast mean, stride 0).
__device__ static inline v16h lds_afrag(const _Float16* base, int stride, int kbase) {
  const int lane  = threadIdx.x & 31;
  const int m     = lane & 15;
  const int khalf = (lane >> 4) << 3;
  const _Float16* row = base + m * stride;
  v16h a;
#pragma unroll
  for (int v = 0; v < 8; ++v) {
    const int k = kbase + ((v & 4) << 2) + khalf + ((v & 3) << 1);
    v2h p = *(const v2h*)(row + k);
    a[2 * v]     = p[0];
    a[2 * v + 1] = p[1];
  }
  return a;
}

// ---- Sequential packed-B stream with one-block software pipelining ----------
struct BStream {
  const _Float16* p;     // lane-offset pointer to current block
  const _Float16* last;  // lane-offset pointer to final block (clamp for lookahead)
  v16h cur;
};
__device__ static inline void bs_init(BStream& s, const _Float16* pk, int nblocks) {
  const int lane = threadIdx.x & 31;
  s.p    = pk + lane * 16;
  s.last = pk + (size_t)(nblocks - 1) * 512 + lane * 16;
  s.cur  = *(const v16h*)s.p;                 // prime the pipeline
}
__device__ static inline v16h bs_next(BStream& s) {
  v16h c = s.cur;
  const _Float16* np = (s.p == s.last) ? s.p : (s.p + 512);
  s.cur = *(const v16h*)np;                   // in flight during the WMMA below
  s.p = np;
  return c;
}

// Accumulator seeded with bias broadcast (D-layout n = lane&15).
__device__ static inline v8f bias_acc(const float* __restrict__ b) {
  const float bv = b[threadIdx.x & 15];
  v8f a;
#pragma unroll
  for (int v = 0; v < 8; ++v) a[v] = bv;
  return a;
}

// Row-major f16 store of a D-tile (for Vt, consumed by the weighted sum).
__device__ static inline void store_tile_f16(_Float16* buf, int stride, int nbase, v8f acc) {
  const int lane = threadIdx.x & 31;
  const int n    = nbase + (lane & 15);
  const int mb   = (lane >> 4) << 3;
#pragma unroll
  for (int v = 0; v < 8; ++v) buf[(v + mb) * stride + n] = (_Float16)acc[v];
}

// Repack fp32 weight W[Kdim][Ndim] into per-block per-lane f16 B-fragments.
// Block order: block = nt*KT + kt  (sequential stream in the GEMM loops).
__global__ void swarm_pack_weight(const float* __restrict__ W, _Float16* __restrict__ out,
                                  int Kdim, int Ndim) {
  const int gid = blockIdx.x * blockDim.x + threadIdx.x;
  const int KT = Kdim >> 5;
  const int nblocks = KT * (Ndim >> 4);
  const int block = gid >> 5;
  if (block >= nblocks) return;
  const int lane  = gid & 31;
  const int kt    = block % KT;
  const int nt    = block / KT;
  const int n     = (nt << 4) + (lane & 15);
  const int khalf = (lane >> 4) << 3;
  _Float16* dst = out + (((size_t)block * 32) + lane) * 16;
#pragma unroll
  for (int v = 0; v < 8; ++v) {
    const int k = (kt << 5) + ((v & 4) << 2) + khalf + ((v & 3) << 1);
    dst[2 * v]     = (_Float16)W[(size_t)k * Ndim + n];
    dst[2 * v + 1] = (_Float16)W[(size_t)(k + 1) * Ndim + n];
  }
}

__launch_bounds__(32)
__global__ void swarm_fused_kernel(
    const float* __restrict__ self_obs, const float* __restrict__ obs,
    const float* __restrict__ be1, const float* __restrict__ be2,
    const float* __restrict__ bv1, const float* __restrict__ bv2,
    const float* __restrict__ ba1, const float* __restrict__ wa2,
    const _Float16* __restrict__ pWe1, const _Float16* __restrict__ pWe2,
    const _Float16* __restrict__ pWv1, const _Float16* __restrict__ pWv2,
    const _Float16* __restrict__ pWa1,
    float* __restrict__ out, int B) {
  __shared__ _Float16 hid[16 * 1024];   // packed A-frag buffer (32 k-tiles max); also X staging
  __shared__ _Float16 Et[16 * 256];     // embeddings, packed A-frag layout (8 k-tiles)
  __shared__ _Float16 Vt[16 * 256];     // values, row-major (weighted sum reads rows)
  __shared__ _Float16 meanh[256];       // per-batch embedding mean
  __shared__ float    lgl[16];          // attention logits

  const int t = blockIdx.x;             // batch element == 16-row tile
  const int lane = threadIdx.x;

  // ---- Stage mlp_input (16 x 128) directly into packed A-frag layout.
  // Row mixing per reference: self from r % B, neighbor from (b = r/16, k = r%16).
#pragma unroll 4
  for (int idx = lane; idx < 16 * 128; idx += 32) {
    const int i = idx >> 7;             // row m (== neighbor k index)
    const int c = idx & 127;            // column k
    const int r = t * 16 + i;
    const float val = (c < 64)
        ? self_obs[(size_t)(r % B) * 64 + c]
        : obs[(size_t)t * 1088 + 64 + i * 64 + (c - 64)];
    const int kt = c >> 5, ko = c & 31, rem = ko & 15;
    const int e  = 2 * (((ko & 16) >> 2) + ((rem & 7) >> 1)) + (rem & 1);
    const int dl = i + ((rem >> 3) << 4);
    hid[(kt * 32 + dl) * 16 + e] = (_Float16)val;
  }
  __syncthreads();

  v16h ax[4];
#pragma unroll
  for (int kt = 0; kt < 4; ++kt) ax[kt] = packed_frag(hid, kt);
  __syncthreads();

  // ---- FF1 layer 1: (16x128)@(128x512) + be1, GELU -> hid packed (16 k-tiles)
  {
    BStream bs; bs_init(bs, pWe1, 4 * 32);
    for (int nt = 0; nt < 32; ++nt) {
      __builtin_prefetch(bs.p + 512 * 8, 0, 1);   // ~2 n-tiles ahead
      v8f acc = bias_acc(be1 + nt * 16);
#pragma unroll
      for (int kt = 0; kt < 4; ++kt) acc = WMMA_F16(ax[kt], bs_next(bs), acc);
      store_packed(hid, nt * 16, acc, true);
    }
  }
  __syncthreads();

  // ---- FF1 layer 2: (16x512)@(512x256) + be2 -> Et packed ; column mean
  {
    v16h ah[16];
#pragma unroll
    for (int kt = 0; kt < 16; ++kt) ah[kt] = packed_frag(hid, kt);
    BStream bs; bs_init(bs, pWe2, 16 * 16);
    for (int nt = 0; nt < 16; ++nt) {
      __builtin_prefetch(bs.p + 512 * 32, 0, 1);
      v8f acc = bias_acc(be2 + nt * 16);
#pragma unroll
      for (int kt = 0; kt < 16; ++kt) acc = WMMA_F16(ah[kt], bs_next(bs), acc);
      store_packed(Et, nt * 16, acc, false);
      float s = 0.f;
#pragma unroll
      for (int v = 0; v < 8; ++v) s += acc[v];    // this lane's 8 rows at n=lane&15
      s += __shfl_xor(s, 16, 32);                 // other half shares n
      if (lane < 16) meanh[nt * 16 + lane] = (_Float16)(s * (1.0f / 16.0f));
    }
  }
  __syncthreads();

  // ---- FFv layer 1: (16x256)@(256x1024) + bv1, GELU -> hid packed (32 k-tiles)
  {
    v16h ae[8];
#pragma unroll
    for (int kt = 0; kt < 8; ++kt) ae[kt] = packed_frag(Et, kt);
    BStream bs; bs_init(bs, pWv1, 8 * 64);
    for (int nt = 0; nt < 64; ++nt) {
      __builtin_prefetch(bs.p + 512 * 16, 0, 1);
      v8f acc = bias_acc(bv1 + nt * 16);
#pragma unroll
      for (int kt = 0; kt < 8; ++kt) acc = WMMA_F16(ae[kt], bs_next(bs), acc);
      store_packed(hid, nt * 16, acc, true);
    }
  }
  __syncthreads();

  // ---- FFv layer 2: (16x1024)@(1024x256) + bv2 -> Vt (row-major)
  {
    v16h ah[32];                                   // 256 VGPRs, wave32 budget is fine
#pragma unroll
    for (int kt = 0; kt < 32; ++kt) ah[kt] = packed_frag(hid, kt);
    BStream bs; bs_init(bs, pWv2, 32 * 16);
    for (int nt = 0; nt < 16; ++nt) {
      __builtin_prefetch(bs.p + 512 * 64, 0, 1);
      v8f acc = bias_acc(bv2 + nt * 16);
#pragma unroll
      for (int kt = 0; kt < 32; ++kt) acc = WMMA_F16(ah[kt], bs_next(bs), acc);
      store_tile_f16(Vt, 256, nt * 16, acc);
    }
  }
  __syncthreads();

  // ---- FFa: attn_in = [E || mean] (16x512)@(512x2048) + ba1, GELU, dot Wa2.
  // The 2048->1 layer is fused into the n-tile loop; ba2 cancels in softmax.
  float lg[8];
#pragma unroll
  for (int v = 0; v < 8; ++v) lg[v] = 0.f;
  {
    v16h aa[16];
#pragma unroll
    for (int kt = 0; kt < 8; ++kt) aa[kt] = packed_frag(Et, kt);
#pragma unroll
    for (int kt = 0; kt < 8; ++kt) aa[8 + kt] = lds_afrag(meanh, 0, kt * 32);
    BStream bs; bs_init(bs, pWa1, 16 * 128);
    for (int nt = 0; nt < 128; ++nt) {
      __builtin_prefetch(bs.p + 512 * 32, 0, 1);
      v8f acc = bias_acc(ba1 + nt * 16);
#pragma unroll
      for (int kt = 0; kt < 16; ++kt) acc = WMMA_F16(aa[kt], bs_next(bs), acc);
      const float w2 = wa2[nt * 16 + (lane & 15)];
#pragma unroll
      for (int v = 0; v < 8; ++v) lg[v] += gelu_exact(acc[v]) * w2;
    }
  }
  // Reduce over the 16 lanes of each half (same m set, different n columns).
#pragma unroll
  for (int off = 1; off < 16; off <<= 1) {
#pragma unroll
    for (int v = 0; v < 8; ++v) lg[v] += __shfl_xor(lg[v], off, 32);
  }
  if ((lane & 15) == 0) {
    const int mb = (lane >> 4) << 3;
#pragma unroll
    for (int v = 0; v < 8; ++v) lgl[mb + v] = lg[v];
  }
  __syncthreads();

  // ---- Softmax over the 16 neighbors + attention-weighted sum of Vt.
  float mx = -3.4e38f;
#pragma unroll
  for (int m = 0; m < 16; ++m) mx = fmaxf(mx, lgl[m]);
  float attn[16];
  float ssum = 0.f;
#pragma unroll
  for (int m = 0; m < 16; ++m) { attn[m] = __expf(lgl[m] - mx); ssum += attn[m]; }
  const float inv = 1.0f / ssum;
#pragma unroll
  for (int jj = 0; jj < 8; ++jj) {
    const int j = jj * 32 + lane;                 // coalesced over lanes
    float a = 0.f;
#pragma unroll
    for (int m = 0; m < 16; ++m) a += attn[m] * (float)Vt[m * 256 + j];
    out[(size_t)t * 256 + j] = a * inv;
  }
}

extern "C" void kernel_launch(void* const* d_in, const int* in_sizes, int n_in,
                              void* d_out, int out_size, void* d_ws, size_t ws_size,
                              hipStream_t stream) {
  (void)n_in; (void)out_size; (void)ws_size;
  const float* self_obs = (const float*)d_in[0];
  const float* obs      = (const float*)d_in[1];
  const float* We1      = (const float*)d_in[2];
  const float* be1      = (const float*)d_in[3];
  const float* We2      = (const float*)d_in[4];
  const float* be2      = (const float*)d_in[5];
  const float* Wv1      = (const float*)d_in[6];
  const float* bv1      = (const float*)d_in[7];
  const float* Wv2      = (const float*)d_in[8];
  const float* bv2      = (const float*)d_in[9];
  const float* Wa1      = (const float*)d_in[10];
  const float* ba1      = (const float*)d_in[11];
  const float* Wa2      = (const float*)d_in[12];
  float* out            = (float*)d_out;

  const int B = in_sizes[0] / 64;  // 8192

  // Workspace: packed f16 weights only (~3.4 MB).
  char* ws = (char*)d_ws;
  _Float16* pWe1 = (_Float16*)(ws + 0);        //  4*32 blocks * 1 KB = 128 KB
  _Float16* pWe2 = (_Float16*)(ws + 131072);   // 16*16 blocks       = 256 KB
  _Float16* pWv1 = (_Float16*)(ws + 393216);   //  8*64 blocks       = 512 KB
  _Float16* pWv2 = (_Float16*)(ws + 917504);   // 32*16 blocks       = 512 KB
  _Float16* pWa1 = (_Float16*)(ws + 1441792);  // 16*128 blocks      =   2 MB

  struct PackJob { const float* W; _Float16* dst; int Kd, Nd; };
  const PackJob jobs[5] = {
      {We1, pWe1, 128, 512},  {We2, pWe2, 512, 256},
      {Wv1, pWv1, 256, 1024}, {Wv2, pWv2, 1024, 256},
      {Wa1, pWa1, 512, 2048},
  };
  for (int i = 0; i < 5; ++i) {
    const int threads = (jobs[i].Kd >> 5) * (jobs[i].Nd >> 4) * 32;
    swarm_pack_weight<<<(threads + 63) / 64, 64, 0, stream>>>(
        jobs[i].W, jobs[i].dst, jobs[i].Kd, jobs[i].Nd);
  }

  swarm_fused_kernel<<<B, 32, 0, stream>>>(
      self_obs, obs, be1, be2, bv1, bv2, ba1, Wa2,
      pWe1, pWe2, pWv1, pWv2, pWa1, out, B);
}
